// SelfAttention_39298950758760
// MI455X (gfx1250) — matrix operand order repaired and checked by reading, not compiled, not run on hardware
//
#include <hip/hip_runtime.h>
#include <hip/hip_bf16.h>

// ---------------------------------------------------------------------------
// Fused causal self-attention + residual + LayerNorm for MI455X (gfx1250).
// bf16 WMMA matmuls, fp32 accumulation, TDM (tensor_load_to_lds) double-
// buffered K/V^T staging in LDS.  B=4, S=4096, D_MODEL=512, D_ATT=64.
// ---------------------------------------------------------------------------

typedef __attribute__((ext_vector_type(16))) __bf16   v16bf;
typedef __attribute__((ext_vector_type(8)))  float    v8f;
typedef __attribute__((ext_vector_type(4)))  unsigned v4u;
typedef __attribute__((ext_vector_type(8)))  int      v8i;
typedef __attribute__((ext_vector_type(4)))  int      v4i;

#define B_    4
#define S_    4096
#define DM_   512
#define DA_   64

union Frag {
    v16bf  v;
    uint4  q[2];
    __bf16 h[16];
};

__device__ __forceinline__ __bf16 tobf(float f) { return (__bf16)f; }

#if __has_builtin(__builtin_amdgcn_tensor_load_to_lds) && \
    __has_builtin(__builtin_amdgcn_s_wait_tensorcnt)
#define HAVE_TDM 1
#else
#define HAVE_TDM 0
#endif

#if HAVE_TDM
// 2-D tile DMA: global -> LDS, bf16 elements (data_size = 2 bytes).
// tensor: [d1 lines x d0 elements], line stride = stride_d0 elements.
// tile:   [tile_d1 lines x tile_d0 elements], packed contiguously in LDS.
__device__ __forceinline__ void tdm_load_2d(
    unsigned lds_off, const void* gptr,
    unsigned tensor_d0, unsigned tensor_d1, unsigned stride_d0,
    unsigned tile_d0, unsigned tile_d1)
{
    unsigned long long ga = (unsigned long long)gptr;
    v4u g0;
    g0[0] = 1u;                                        // count=1, user D#
    g0[1] = lds_off;                                   // lds_addr (bytes)
    g0[2] = (unsigned)(ga & 0xffffffffu);              // global_addr[31:0]
    g0[3] = (unsigned)((ga >> 32) & 0x01ffffffu)       // global_addr[56:32]
          | (2u << 30);                                // type = 2 ("image")
    v8i g1;
    g1[0] = (int)(1u << 16);                           // wg_mask=0, data_size=2B
    g1[1] = (int)((tensor_d0 & 0xffffu) << 16);        // tensor_dim0[15:0]
    g1[2] = (int)((tensor_d0 >> 16) & 0xffffu)         // tensor_dim0[31:16]
          | (int)((tensor_d1 & 0xffffu) << 16);        // tensor_dim1[15:0]
    g1[3] = (int)((tensor_d1 >> 16) & 0xffffu)         // tensor_dim1[31:16]
          | (int)(tile_d0 << 16);                      // tile_dim0
    g1[4] = (int)(tile_d1 & 0xffffu);                  // tile_dim1 (tile_dim2=0)
    g1[5] = (int)stride_d0;                            // tensor_dim0_stride[31:0]
    g1[6] = 0;                                         // stride hi bits = 0
    g1[7] = 0;
    v4i g2 = {};                                       // 2-D: groups 2/3 unused
    v4i g3 = {};
#if defined(__clang_major__) && (__clang_major__ >= 23)
    v8i g4 = {};
    __builtin_amdgcn_tensor_load_to_lds(g0, g1, g2, g3, g4, 0);
#else
    __builtin_amdgcn_tensor_load_to_lds(g0, g1, g2, g3, 0);
#endif
}
#endif

// ---------------------------------------------------------------------------
// Projection GEMM: out(bf16) = x[16384,512] @ W[512,C] + bias.
// One wave -> 16x64 tile. W col-slice staged in LDS as W^T[n][k] bf16.
// transposed==0 : out is [B*S, C] row-major     (Q, K;  C==64)
// transposed==1 : out is [B, C, S] (V^T)        (V;     C==512)
// ---------------------------------------------------------------------------
__global__ __launch_bounds__(128) void proj_kernel(
    const float* __restrict__ x, const float* __restrict__ W,
    const float* __restrict__ bias, __bf16* __restrict__ out,
    int C, int transposed)
{
    __shared__ __bf16 wlds[64 * 512];   // 64 KB: W^T slice, [n][k] bf16

    const int tid  = threadIdx.x;
    const int wid  = tid >> 5;
    const int lane = tid & 31;
    const int m    = lane & 15;
    const int half = lane >> 4;
    const int n0   = blockIdx.y * 64;

    for (int idx = tid; idx < 64 * 512; idx += 128) {
        int n = idx & 63;
        int k = idx >> 6;
        wlds[n * 512 + k] = tobf(W[(size_t)k * C + n0 + n]);
    }
    __syncthreads();

    const int row0 = blockIdx.x * 64 + wid * 16;
    const float* xrow = x + (size_t)(row0 + m) * DM_;

    v8f acc[4] = {};

    for (int k = 0; k < 512; k += 32) {
        Frag a;
#pragma unroll
        for (int c = 0; c < 2; ++c) {
            int off = k + c * 16 + half * 8;
            float4 f0 = *(const float4*)(xrow + off);
            float4 f1 = *(const float4*)(xrow + off + 4);
            a.h[c * 8 + 0] = tobf(f0.x);
            a.h[c * 8 + 1] = tobf(f0.y);
            a.h[c * 8 + 2] = tobf(f0.z);
            a.h[c * 8 + 3] = tobf(f0.w);
            a.h[c * 8 + 4] = tobf(f1.x);
            a.h[c * 8 + 5] = tobf(f1.y);
            a.h[c * 8 + 6] = tobf(f1.z);
            a.h[c * 8 + 7] = tobf(f1.w);
        }
#pragma unroll
        for (int t = 0; t < 4; ++t) {
            const __bf16* bp = &wlds[(t * 16 + m) * 512 + k + half * 8];
            Frag bf;
            bf.q[0] = *(const uint4*)bp;
            bf.q[1] = *(const uint4*)(bp + 16);
            acc[t] = __builtin_amdgcn_wmma_f32_16x16x32_bf16(
                false, a.v, false, bf.v, (short)0, acc[t], false, false);
        }
    }

    if (!transposed) {
#pragma unroll
        for (int t = 0; t < 4; ++t) {
            int col = n0 + t * 16 + m;
            float bv = bias[col];
#pragma unroll
            for (int r = 0; r < 8; ++r) {
                int row = row0 + r + half * 8;
                out[(size_t)row * C + col] = tobf(acc[t][r] + bv);
            }
        }
    } else {
        const int b  = row0 >> 12;        // S_ == 4096
        const int s0 = row0 & (S_ - 1);
#pragma unroll
        for (int t = 0; t < 4; ++t) {
            int col = n0 + t * 16 + m;
            float bv = bias[col];
#pragma unroll
            for (int r = 0; r < 8; ++r) {
                int s = s0 + r + half * 8;
                out[((size_t)(b * DM_ + col)) * S_ + s] = tobf(acc[t][r] + bv);
            }
        }
    }
}

// ---------------------------------------------------------------------------
// Flash attention + residual + LayerNorm.
// 1 wave = 16 queries, 4 waves/block, 256 blocks.  All waves share a block-
// uniform 64-key KV loop (same trip count for every wave), so K[64x64] and
// V^T[512x64] tiles are staged in LDS by the Tensor Data Mover, double-
// buffered, and consumed by 72 WMMAs per step.
// ---------------------------------------------------------------------------
__global__ __launch_bounds__(128) void attn_kernel(
    const float* __restrict__ x,
    const __bf16* __restrict__ qb,   // [B*S, 64]  bf16
    const __bf16* __restrict__ kb,   // [B*S, 64]  bf16
    const __bf16* __restrict__ vT,   // [B, 512, S] bf16
    const float* __restrict__ gamma, const float* __restrict__ beta,
    float* __restrict__ out)
{
    __shared__ __bf16 kTile[2][64 * 64];    //  16 KB: [key][d]
    __shared__ __bf16 vTile[2][512 * 64];   // 128 KB: [n][key]
    __shared__ __bf16 plds[4][16 * 64];     //   8 KB: per-wave P staging

    const int tid  = threadIdx.x;
    const int wid  = tid >> 5;
    const int lane = tid & 31;
    const int m    = lane & 15;
    const int half = lane >> 4;

    const int qg0    = blockIdx.x * 64;             // global row base of block
    const int b      = qg0 >> 12;                   // batch
    const int qblock = qg0 & (S_ - 1);              // in-batch base of block
    const int qbase  = qblock + wid * 16;           // in-batch base of wave

    const __bf16* kbB = kb + (size_t)b * S_ * DA_;
    const __bf16* vTb = vT + (size_t)b * DM_ * S_;

    // Q A-fragments (d = 0..31, 32..63)
    Frag qa[2];
    {
        const __bf16* qp = qb + (size_t)(qg0 + wid * 16 + m) * DA_;
#pragma unroll
        for (int j = 0; j < 2; ++j) {
            qa[j].q[0] = *(const uint4*)(qp + j * 32 + half * 8);
            qa[j].q[1] = *(const uint4*)(qp + j * 32 + 16 + half * 8);
        }
    }

    float g[32], be[32];
#pragma unroll
    for (int t = 0; t < 32; ++t) {
        g[t]  = gamma[t * 16 + m];
        be[t] = beta[t * 16 + m];
    }

    float ms[8], ls[8];
#pragma unroll
    for (int r = 0; r < 8; ++r) { ms[r] = -1e30f; ls[r] = 0.0f; }
    v8f acc[32] = {};                               // O tile 16 x 512 (fp32)

    const float L2E = 1.4426950408889634f;
    const int nsteps = (qblock >> 6) + 1;           // identical for all waves

#if HAVE_TDM
    if (wid == 0) {   // TDM ignores EXEC; one issuing wave per block
        tdm_load_2d((unsigned)(uintptr_t)&kTile[0][0], kbB,
                    DA_, S_, DA_, 64, 64);
        tdm_load_2d((unsigned)(uintptr_t)&vTile[0][0], vTb,
                    S_, DM_, S_, 64, 512);
    }
#endif

    for (int i = 0; i < nsteps; ++i) {
        const int k0  = i << 6;
        const int cur = i & 1;

#if HAVE_TDM
        if (wid == 0) {
            if (i + 1 < nsteps) {   // prefetch next step into other buffer
                const int nk0 = (i + 1) << 6;
                tdm_load_2d((unsigned)(uintptr_t)&kTile[cur ^ 1][0],
                            kbB + (size_t)nk0 * DA_, DA_, S_, DA_, 64, 64);
                tdm_load_2d((unsigned)(uintptr_t)&vTile[cur ^ 1][0],
                            vTb + nk0, S_, DM_, S_, 64, 512);
                __builtin_amdgcn_s_wait_tensorcnt(2);  // step i retired
            } else {
                __builtin_amdgcn_s_wait_tensorcnt(0);
            }
        }
#else
        {   // fallback: cooperative global->LDS staging of step i
            for (int idx = tid; idx < 64 * 64 / 8; idx += 128) {
                int key = idx >> 3, c = idx & 7;
                *(uint4*)&kTile[cur][key * 64 + c * 8] =
                    *(const uint4*)(kbB + (size_t)(k0 + key) * DA_ + c * 8);
            }
            for (int idx = tid; idx < 512 * 64 / 8; idx += 128) {
                int n = idx >> 3, c = idx & 7;
                *(uint4*)&vTile[cur][n * 64 + c * 8] =
                    *(const uint4*)(vTb + (size_t)n * S_ + k0 + c * 8);
            }
        }
#endif
        __syncthreads();   // tile[cur] visible to all waves

        // ---- scores: 16 x 64 (four 16x16 C tiles) ------------------------
        v8f cc[4] = {};
#pragma unroll
        for (int kt = 0; kt < 4; ++kt) {
            const __bf16* kp = &kTile[cur][(kt * 16 + m) * 64 + half * 8];
            Frag k0f, k1f;                       // d 0..31 and 32..63
            k0f.q[0] = *(const uint4*)(kp);
            k0f.q[1] = *(const uint4*)(kp + 16);
            k1f.q[0] = *(const uint4*)(kp + 32);
            k1f.q[1] = *(const uint4*)(kp + 48);
            cc[kt] = __builtin_amdgcn_wmma_f32_16x16x32_bf16(
                false, qa[0].v, false, k0f.v, (short)0, cc[kt], false, false);
            cc[kt] = __builtin_amdgcn_wmma_f32_16x16x32_bf16(
                false, qa[1].v, false, k1f.v, (short)0, cc[kt], false, false);
        }

        // ---- causal mask + online softmax --------------------------------
#pragma unroll
        for (int r = 0; r < 8; ++r) {
            const int row  = qbase + r + half * 8;
            const int col0 = k0 + m;
            float s0v = (col0      > row) ? -1e30f : cc[0][r];
            float s1v = (col0 + 16 > row) ? -1e30f : cc[1][r];
            float s2v = (col0 + 32 > row) ? -1e30f : cc[2][r];
            float s3v = (col0 + 48 > row) ? -1e30f : cc[3][r];

            float mx = fmaxf(fmaxf(s0v, s1v), fmaxf(s2v, s3v));
            mx = fmaxf(mx, __shfl_xor(mx, 1, 32));
            mx = fmaxf(mx, __shfl_xor(mx, 2, 32));
            mx = fmaxf(mx, __shfl_xor(mx, 4, 32));
            mx = fmaxf(mx, __shfl_xor(mx, 8, 32));
            const float mnew  = fmaxf(ms[r], mx);
            const float p0    = exp2f((s0v - mnew) * L2E);
            const float p1    = exp2f((s1v - mnew) * L2E);
            const float p2    = exp2f((s2v - mnew) * L2E);
            const float p3    = exp2f((s3v - mnew) * L2E);
            const float alpha = exp2f((ms[r] - mnew) * L2E);
            ms[r] = mnew;

            float rs = (p0 + p1) + (p2 + p3);
            rs += __shfl_xor(rs, 1, 32);
            rs += __shfl_xor(rs, 2, 32);
            rs += __shfl_xor(rs, 4, 32);
            rs += __shfl_xor(rs, 8, 32);
            ls[r] = ls[r] * alpha + rs;

            __bf16* pr = &plds[wid][(r + half * 8) * 64];
            pr[m]      = tobf(p0);
            pr[m + 16] = tobf(p1);
            pr[m + 32] = tobf(p2);
            pr[m + 48] = tobf(p3);

#pragma unroll
            for (int t = 0; t < 32; ++t) acc[t][r] *= alpha;
        }

        // ---- P A-fragments from LDS (keys 0..31, 32..63) -----------------
        Frag pf[2];
#pragma unroll
        for (int j = 0; j < 2; ++j) {
            const __bf16* pp = &plds[wid][m * 64 + j * 32 + half * 8];
            pf[j].q[0] = *(const uint4*)pp;
            pf[j].q[1] = *(const uint4*)(pp + 16);
        }

        // ---- O += P @ V : 32 column tiles x 2 key halves -----------------
#pragma unroll
        for (int t = 0; t < 32; ++t) {
            const __bf16* vp = &vTile[cur][(t * 16 + m) * 64 + half * 8];
            Frag vf0, vf1;
            vf0.q[0] = *(const uint4*)vp;
            vf0.q[1] = *(const uint4*)(vp + 16);
            vf1.q[0] = *(const uint4*)(vp + 32);
            vf1.q[1] = *(const uint4*)(vp + 48);
            acc[t] = __builtin_amdgcn_wmma_f32_16x16x32_bf16(
                false, pf[0].v, false, vf0.v, (short)0, acc[t], false, false);
            acc[t] = __builtin_amdgcn_wmma_f32_16x16x32_bf16(
                false, pf[1].v, false, vf1.v, (short)0, acc[t], false, false);
        }

        __syncthreads();   // all waves done with tile[cur] before overwrite
    }

    // ---- epilogue: 1/l, residual, LayerNorm, store -----------------------
#pragma unroll
    for (int r = 0; r < 8; ++r) {
        const int    srow = qbase + r + half * 8;
        const size_t row  = (size_t)b * S_ + srow;
        const float* xr   = x + row * DM_;
        const float  inv  = 1.0f / ls[r];

        float sum = 0.0f;
#pragma unroll
        for (int t = 0; t < 32; ++t) {
            float y = acc[t][r] * inv + xr[t * 16 + m];
            acc[t][r] = y;
            sum += y;
        }
        sum += __shfl_xor(sum, 1, 32);
        sum += __shfl_xor(sum, 2, 32);
        sum += __shfl_xor(sum, 4, 32);
        sum += __shfl_xor(sum, 8, 32);
        const float mu = sum * (1.0f / 512.0f);

        float vs = 0.0f;
#pragma unroll
        for (int t = 0; t < 32; ++t) {
            float d = acc[t][r] - mu;
            vs += d * d;
        }
        vs += __shfl_xor(vs, 1, 32);
        vs += __shfl_xor(vs, 2, 32);
        vs += __shfl_xor(vs, 4, 32);
        vs += __shfl_xor(vs, 8, 32);
        const float rstd = rsqrtf(vs * (1.0f / 512.0f) + 1e-5f);

        float* orow = out + row * DM_;
#pragma unroll
        for (int t = 0; t < 32; ++t)
            orow[t * 16 + m] = (acc[t][r] - mu) * rstd * g[t] + be[t];
    }
}

// ---------------------------------------------------------------------------
extern "C" void kernel_launch(void* const* d_in, const int* in_sizes, int n_in,
                              void* d_out, int out_size, void* d_ws, size_t ws_size,
                              hipStream_t stream) {
    (void)in_sizes; (void)n_in; (void)out_size; (void)ws_size;

    const float* x     = (const float*)d_in[0];
    const float* Wq    = (const float*)d_in[1];
    const float* bq    = (const float*)d_in[2];
    const float* Wk    = (const float*)d_in[3];
    const float* bk    = (const float*)d_in[4];
    const float* Wv    = (const float*)d_in[5];
    const float* bv    = (const float*)d_in[6];
    const float* gamma = (const float*)d_in[7];
    const float* beta  = (const float*)d_in[8];

    // workspace: Q (2 MB) | K (2 MB) | V^T (16 MB), all bf16
    __bf16* qws = (__bf16*)d_ws;
    __bf16* kws = qws + (size_t)B_ * S_ * DA_;
    __bf16* vws = kws + (size_t)B_ * S_ * DA_;

    const int rowBlocks = (B_ * S_) / 64;   // 256

    proj_kernel<<<dim3(rowBlocks, 1), 128, 0, stream>>>(x, Wq, bq, qws, DA_, 0);
    proj_kernel<<<dim3(rowBlocks, 1), 128, 0, stream>>>(x, Wk, bk, kws, DA_, 0);
    proj_kernel<<<dim3(rowBlocks, DM_ / 64), 128, 0, stream>>>(x, Wv, bv, vws, DM_, 1);
    attn_kernel<<<rowBlocks, 128, 0, stream>>>(x, qws, kws, vws, gamma, beta,
                                               (float*)d_out);
}
